// GE2ELoss_11527692223241
// MI455X (gfx1250) — matrix operand order repaired and checked by reading, not compiled, not run on hardware
//
#include <hip/hip_runtime.h>
#include <hip/hip_bf16.h>
#include <math.h>

// ---------------------------------------------------------------------------
// GE2E loss for gfx1250 (MI455X), fused:
//   G = dvecs(32768x512) * S^T(512x1024) via bf16-split WMMA (hi*hi+hi*lo+lo*hi)
//   then per-row log-softmax with leave-one-out diagonal, all fp32.
// Tiling: 1 WG (8 wave32) = 32 rows (one speaker) x 1024 cols.
//   wave = 2 row-stripes x 8 col-tiles -> each B fragment feeds 6 WMMAs.
// Split-S stored K-chunk-major [c][n][32], Stl contiguous after Sth so every
// B load is one base pointer + constant immediate offset (no per-tile addrs).
// ---------------------------------------------------------------------------

typedef __attribute__((ext_vector_type(8)))  __bf16 v8bf;
typedef __attribute__((ext_vector_type(16))) __bf16 v16bf;
typedef __attribute__((ext_vector_type(8)))  float  v8f;

#define N_SPK 1024
#define M_UTT 32
#define D_DIM 512
#define NROWS (N_SPK * M_UTT)    // 32768
#define NCHUNK (D_DIM / 32)      // 16 K-chunks
#define ROWH 520                 // LDS A row stride in halfs (512 + 8 pad)
#define AOF_L (32 * ROWH)        // Ahi -> Alo delta (halfs)
#define AOF_S (16 * ROWH)        // stripe delta (halfs)
#define STL_DELTA (N_SPK * D_DIM)  // Sth -> Stl delta in halfs (1 MB)
#define SMEM_A (2 * 32 * ROWH * 2)             // Ahi+Alo bytes = 66560
#define SMEM_BYTES (SMEM_A + (160 + 256) * 4)  // + reduction scratch = 68224

// ---- Kernel 1: per-speaker sums S_k -> bf16 hi/lo split (transposed) + |S_k|^2
__global__ __launch_bounds__(256) void k_centroid(const float* __restrict__ dvecs,
                                                  __bf16* __restrict__ Sth,
                                                  float* __restrict__ s_norm2) {
  __shared__ float red[8];
  const int k   = blockIdx.x;
  const int tid = threadIdx.x;
  const float* base = dvecs + (size_t)k * (M_UTT * D_DIM);
  float ssq = 0.f;
  for (int d = tid; d < D_DIM; d += 256) {
    float s = 0.f;
#pragma unroll
    for (int m = 0; m < M_UTT; ++m) s += base[m * D_DIM + d];
    __bf16 h = (__bf16)s;
    __bf16 l = (__bf16)(s - (float)h);
    const int c = d >> 5, k32 = d & 31;
    const size_t off = ((size_t)c * N_SPK + k) * 32 + k32;  // [c][n][32]
    Sth[off] = h;
    Sth[STL_DELTA + off] = l;
    ssq += s * s;
  }
#pragma unroll
  for (int m = 16; m >= 1; m >>= 1) ssq += __shfl_xor(ssq, m);
  if ((tid & 31) == 0) red[tid >> 5] = ssq;
  __syncthreads();
  if (tid == 0) {
    float t = 0.f;
#pragma unroll
    for (int w = 0; w < 8; ++w) t += red[w];
    s_norm2[k] = t;
  }
}

// ---- Kernel 2: per-utterance |e|^2 (one wave per row) ---------------------
__global__ __launch_bounds__(256) void k_enorm(const float* __restrict__ dvecs,
                                               float* __restrict__ e_norm2) {
  const int row  = blockIdx.x * 8 + (threadIdx.x >> 5);
  const int lane = threadIdx.x & 31;
  const float4* p = (const float4*)(dvecs + (size_t)row * D_DIM);
  float s = 0.f;
#pragma unroll
  for (int c = 0; c < 4; ++c) {
    float4 v = p[c * 32 + lane];
    s += v.x * v.x + v.y * v.y + v.z * v.z + v.w * v.w;
  }
#pragma unroll
  for (int m = 16; m >= 1; m >>= 1) s += __shfl_xor(s, m);
  if (lane == 0) e_norm2[row] = s;
}

// ---- Kernel 3: fused GEMM + clamp/affine + LOO diagonal + log-softmax -----
__global__ __launch_bounds__(256) void k_main(const float* __restrict__ dvecs,
                                              const __bf16* __restrict__ Sth,
                                              const float* __restrict__ s_norm2,
                                              const float* __restrict__ e_norm2,
                                              const float* __restrict__ wptr,
                                              const float* __restrict__ bptr,
                                              float* __restrict__ block_loss) {
  extern __shared__ __align__(16) char smem[];
  __bf16* Ahi = (__bf16*)smem;                  // [32][ROWH], Alo follows at AOF_L
  float* scr     = (float*)(smem + SMEM_A);
  float* en2S    = scr;          // [32]
  float* einvS   = scr + 32;     // [32]
  float* rowmax  = scr + 64;     // [32]
  float* rowdiag = scr + 96;     // [32]
  float* rowloss = scr + 128;    // [32]
  float (*wred)[8] = (float(*)[8])(scr + 160);  // [32][8]

  const int tid  = threadIdx.x;
  const int jspk = blockIdx.x;      // this WG = speaker jspk
  const int jm0  = jspk * 32;       // 32 global rows
  const float W = *wptr, B = *bptr;

  // ---- fill LDS with bf16 hi/lo split of the 32x512 A block ----
  {
    const float4* src = (const float4*)(dvecs + (size_t)jm0 * D_DIM);
    for (int i = tid; i < 32 * 128; i += 256) {
      int row = i >> 7;
      int pos = (i & 127) << 2;
      float4 v = src[i];
      float vv[4] = {v.x, v.y, v.z, v.w};
#pragma unroll
      for (int e = 0; e < 4; ++e) {
        __bf16 h = (__bf16)vv[e];
        __bf16 l = (__bf16)(vv[e] - (float)h);
        Ahi[row * ROWH + pos + e] = h;
        Ahi[AOF_L + row * ROWH + pos + e] = l;
      }
    }
    if (tid < 32) {
      float e2 = e_norm2[jm0 + tid];
      en2S[tid]  = e2;
      einvS[tid] = rsqrtf(e2);
    }
  }
  __syncthreads();

  const int wv   = tid >> 5;
  const int lane = tid & 31;
  const int ln   = lane & 15;     // col-within-tile / A-row selector
  const int lhi  = lane >> 4;     // half-wave selector

  v8f acc0[8] = {};   // stripe 0 (rows 0..15)
  v8f acc1[8] = {};   // stripe 1 (rows 16..31)

  // Single LDS base; all 8 A-fragment reads are base + constant offset.
  const __bf16* ab = Ahi + ln * ROWH + lhi * 8;
  // Single global B base; all 32 B loads per chunk are base + constant offset:
  //   tile t -> + t*512 halfs, lo array -> + STL_DELTA halfs.
  const __bf16* bb = Sth + (size_t)(wv * 4096 + ln * 32 + lhi * 8);

#pragma unroll 1
  for (int c = 0; c < NCHUNK; ++c) {
    v8bf q0, q1;
    q0 = *(const v8bf*)(ab);                 q1 = *(const v8bf*)(ab + 16);
    v16bf Ah0 = __builtin_shufflevector(q0, q1, 0,1,2,3,4,5,6,7,8,9,10,11,12,13,14,15);
    q0 = *(const v8bf*)(ab + AOF_L);         q1 = *(const v8bf*)(ab + AOF_L + 16);
    v16bf Al0 = __builtin_shufflevector(q0, q1, 0,1,2,3,4,5,6,7,8,9,10,11,12,13,14,15);
    q0 = *(const v8bf*)(ab + AOF_S);         q1 = *(const v8bf*)(ab + AOF_S + 16);
    v16bf Ah1 = __builtin_shufflevector(q0, q1, 0,1,2,3,4,5,6,7,8,9,10,11,12,13,14,15);
    q0 = *(const v8bf*)(ab + AOF_L + AOF_S); q1 = *(const v8bf*)(ab + AOF_L + AOF_S + 16);
    v16bf Al1 = __builtin_shufflevector(q0, q1, 0,1,2,3,4,5,6,7,8,9,10,11,12,13,14,15);

#pragma unroll
    for (int t = 0; t < 8; ++t) {
      const __bf16* p = bb + t * 512;
      v8bf b0 = *(const v8bf*)(p);
      v8bf b1 = *(const v8bf*)(p + 16);
      v16bf Bh = __builtin_shufflevector(b0, b1, 0,1,2,3,4,5,6,7,8,9,10,11,12,13,14,15);
      v8bf b2 = *(const v8bf*)(p + STL_DELTA);
      v8bf b3 = *(const v8bf*)(p + STL_DELTA + 16);
      v16bf Bl = __builtin_shufflevector(b2, b3, 0,1,2,3,4,5,6,7,8,9,10,11,12,13,14,15);
      acc0[t] = __builtin_amdgcn_wmma_f32_16x16x32_bf16(false, Ah0, false, Bh, (short)0, acc0[t], false, false);
      acc0[t] = __builtin_amdgcn_wmma_f32_16x16x32_bf16(false, Ah0, false, Bl, (short)0, acc0[t], false, false);
      acc0[t] = __builtin_amdgcn_wmma_f32_16x16x32_bf16(false, Al0, false, Bh, (short)0, acc0[t], false, false);
      acc1[t] = __builtin_amdgcn_wmma_f32_16x16x32_bf16(false, Ah1, false, Bh, (short)0, acc1[t], false, false);
      acc1[t] = __builtin_amdgcn_wmma_f32_16x16x32_bf16(false, Ah1, false, Bl, (short)0, acc1[t], false, false);
      acc1[t] = __builtin_amdgcn_wmma_f32_16x16x32_bf16(false, Al1, false, Bh, (short)0, acc1[t], false, false);
    }
    ab += 32;                 // next K-chunk in LDS (+64 B)
    bb += N_SPK * 32;         // next K-chunk of split-S (+64 KB)
  }

  // Per-tile column properties (loaded AFTER the GEMM to keep pressure low).
  int   kcol[8];
  float sinv[8], s2l[8];
#pragma unroll
  for (int t = 0; t < 8; ++t) {
    int k = (wv * 8 + t) * 16 + ln;
    kcol[t] = k;
    float s2 = s_norm2[k];
    s2l[t]  = s2;
    sinv[t] = rsqrtf(s2);
  }

  // C layout: VGPR r holds M = r + 8*lhi (per stripe), N = kcol[t].
  auto xform = [&](const v8f* acc, int s, int t, int r) -> float {
    float g = acc[t][r];
    const int row = s * 16 + r + lhi * 8;
    float cosv;
    if (kcol[t] == jspk) {
      // leave-one-out: (g - |e|^2) / (|e| * sqrt(|S|^2 - 2g + |e|^2))
      float e2 = en2S[row];
      cosv = (g - e2) * einvS[row] * rsqrtf(fmaxf(s2l[t] - 2.f * g + e2, 1e-20f));
    } else {
      cosv = g * einvS[row] * sinv[t];
    }
    return fmaxf(cosv, 1e-6f) * W + B;
  };

  // ---- phase A: deterministic row max (+ capture diagonal logits) ----
  float lmax[2][8];
#pragma unroll
  for (int s = 0; s < 2; ++s)
#pragma unroll
    for (int r = 0; r < 8; ++r) lmax[s][r] = -3.4e38f;
#pragma unroll
  for (int t = 0; t < 8; ++t) {
    const bool isdiag = (kcol[t] == jspk);
#pragma unroll
    for (int s = 0; s < 2; ++s) {
      const v8f* acc = s ? acc1 : acc0;
#pragma unroll
      for (int r = 0; r < 8; ++r) {
        float x = xform(acc, s, t, r);
        lmax[s][r] = fmaxf(lmax[s][r], x);
        if (isdiag) rowdiag[s * 16 + r + lhi * 8] = x;
      }
    }
  }
#pragma unroll
  for (int s = 0; s < 2; ++s)
#pragma unroll
    for (int r = 0; r < 8; ++r) {
      float v = lmax[s][r];
      v = fmaxf(v, __shfl_xor(v, 1));
      v = fmaxf(v, __shfl_xor(v, 2));
      v = fmaxf(v, __shfl_xor(v, 4));
      v = fmaxf(v, __shfl_xor(v, 8));
      if (ln == 0) wred[s * 16 + r + lhi * 8][wv] = v;
    }
  __syncthreads();
  if (tid < 32) {
    float v = wred[tid][0];
#pragma unroll
    for (int w = 1; w < 8; ++w) v = fmaxf(v, wred[tid][w]);
    rowmax[tid] = v;
  }
  __syncthreads();

  // ---- phase B: deterministic sum of exp(x - rowmax) ----
  float rmax_r[2][8];
#pragma unroll
  for (int s = 0; s < 2; ++s)
#pragma unroll
    for (int r = 0; r < 8; ++r) rmax_r[s][r] = rowmax[s * 16 + r + lhi * 8];
  float psum[2][8] = {};
#pragma unroll
  for (int t = 0; t < 8; ++t)
#pragma unroll
    for (int s = 0; s < 2; ++s) {
      const v8f* acc = s ? acc1 : acc0;
#pragma unroll
      for (int r = 0; r < 8; ++r)
        psum[s][r] += __expf(xform(acc, s, t, r) - rmax_r[s][r]);
    }
#pragma unroll
  for (int s = 0; s < 2; ++s)
#pragma unroll
    for (int r = 0; r < 8; ++r) {
      float v = psum[s][r];
      v += __shfl_xor(v, 1);
      v += __shfl_xor(v, 2);
      v += __shfl_xor(v, 4);
      v += __shfl_xor(v, 8);
      if (ln == 0) wred[s * 16 + r + lhi * 8][wv] = v;
    }
  __syncthreads();
  if (tid < 32) {
    float sum = 0.f;
#pragma unroll
    for (int w = 0; w < 8; ++w) sum += wred[tid][w];
    rowloss[tid] = logf(sum) + rowmax[tid] - rowdiag[tid];
  }
  __syncthreads();
  if (tid == 0) {
    float tot = 0.f;
#pragma unroll
    for (int i = 0; i < 32; ++i) tot += rowloss[i];
    block_loss[jspk] = tot;
  }
}

// ---- Kernel 4: fixed-order final reduction (also initializes d_out) -------
__global__ __launch_bounds__(256) void k_final(const float* __restrict__ block_loss,
                                               float* __restrict__ out) {
  __shared__ float red[256];
  const int tid = threadIdx.x;
  float s = 0.f;
  for (int i = tid; i < N_SPK; i += 256) s += block_loss[i];
  red[tid] = s;
  __syncthreads();
  for (int k = 128; k >= 1; k >>= 1) {
    if (tid < k) red[tid] += red[tid + k];
    __syncthreads();
  }
  if (tid == 0) out[0] = red[0];
}

extern "C" void kernel_launch(void* const* d_in, const int* in_sizes, int n_in,
                              void* d_out, int out_size, void* d_ws, size_t ws_size,
                              hipStream_t stream) {
  (void)in_sizes; (void)n_in; (void)out_size; (void)ws_size;
  const float* dvecs = (const float*)d_in[0];
  const float* w     = (const float*)d_in[1];
  const float* b     = (const float*)d_in[2];
  float* out = (float*)d_out;

  // Workspace (< 2.2 MB): Sth(+Stl contiguous) | s_norm2 | e_norm2 | block_loss
  __bf16* Sth = (__bf16*)d_ws;                       // [2][16][1024][32] halfs
  float* s_norm2    = (float*)(Sth + (size_t)2 * N_SPK * D_DIM);
  float* e_norm2    = s_norm2 + N_SPK;
  float* block_loss = e_norm2 + NROWS;

  k_centroid<<<N_SPK, 256, 0, stream>>>(dvecs, Sth, s_norm2);
  k_enorm<<<NROWS / 8, 256, 0, stream>>>(dvecs, e_norm2);
  k_main<<<N_SPK, 256, SMEM_BYTES, stream>>>(dvecs, Sth, s_norm2, e_norm2, w, b, block_loss);
  k_final<<<1, 256, 0, stream>>>(block_loss, out);
}
